// RNN_87668872446047
// MI455X (gfx1250) — matrix operand order repaired and checked by reading, not compiled, not run on hardware
//
#include <hip/hip_runtime.h>
#include <hip/hip_bf16.h>
#include <cstdint>

typedef _Float16 h16;
typedef __attribute__((ext_vector_type(16))) _Float16 v16h;
typedef __attribute__((ext_vector_type(8)))  _Float16 v8h;
typedef __attribute__((ext_vector_type(8)))  float    v8f;

constexpr int Tn = 512, Bn = 64, Vn = 256, Hn = 1024, G = 4096; // G = 4*H

// load a v16h WMMA fragment: 8 halfs at p, 8 halfs at p+off
__device__ inline v16h ldfrag(const h16* __restrict__ p, int off) {
  v8h lo = *(const v8h*)p;
  v8h hi = *(const v8h*)(p + off);
  return __builtin_shufflevector(lo, hi, 0,1,2,3,4,5,6,7,8,9,10,11,12,13,14,15);
}

__device__ inline float sigf(float x) { return 1.f / (1.f + __expf(-x)); }

// ---------------- prep kernels ----------------

__global__ void k_tokens(const float* __restrict__ in, int* __restrict__ tok, int n) {
  int i = blockIdx.x*256 + threadIdx.x;
  if (i >= n) return;
  const float* r = in + (size_t)i*Vn;
  int t = 0;
  for (int v = 0; v < Vn; ++v) if (r[v] > 0.5f) t = v;
  tok[i] = t;
}

__global__ void k_conv(const float* __restrict__ s, h16* __restrict__ d, int n) {
  int i = blockIdx.x*256 + threadIdx.x;
  if (i < n) d[i] = (h16)s[i];
}

// Pre-swizzle fp32 weight [J rows = N][I cols = K] into WMMA-B fragment order:
// dst[((kc*(J/16) + jtile)*32 + lane)*16 + e] = W[jtile*16 + (lane&15)][kc*32 + (lane>>4)*16 + e]
// so a lane's 16 halfs (K-contiguous for its column) are 32B-contiguous in memory.
__global__ void k_swz(const float* __restrict__ src, h16* __restrict__ dst, int J, int I) {
  int idx = blockIdx.x*256 + threadIdx.x;
  int jt_n = J >> 4;
  int total = (I >> 5) * jt_n * 32;
  if (idx >= total) return;
  int lane  = idx & 31;
  int jtile = (idx >> 5) % jt_n;
  int kc    = (idx >> 5) / jt_n;
  int Nrow  = jtile*16 + (lane & 15);
  int K0    = kc*32 + (lane >> 4)*16;
  const float* s = src + (size_t)Nrow*I + K0;
  h16* d = dst + ((size_t)idx) * 16;
  #pragma unroll
  for (int e = 0; e < 16; ++e) d[e] = (h16)s[e];
}

__global__ void k_bsum(const float* __restrict__ b0a, const float* __restrict__ b0b,
                       const float* __restrict__ bra, const float* __restrict__ brb,
                       float* __restrict__ bsum) {
  int i = blockIdx.x*256 + threadIdx.x;
  if (i >= 3*G) return;
  int l = i >> 12, j = i & (G-1);
  bsum[i] = (l == 0) ? (b0a[j] + b0b[j]) : (bra[(l-1)*G + j] + brb[(l-1)*G + j]);
}

__global__ void k_zero(h16* __restrict__ hh, float* __restrict__ cst, int nh, int nc) {
  int i = blockIdx.x*256 + threadIdx.x;
  if (i < nh) hh[i] = (h16)0.f;
  if (i < nc) cst[i] = 0.f;
}

// ---------------- wavefront step kernel (no LDS, no barriers) ----------------
// Launch k: layer0 -> t=k, layer1 -> t=k-1, layer2 -> t=k-2, projection -> t=k-3.
// Blocks 0..47: 3 layers x 16 hidden-slices (64 units, all 4 gates).
// Blocks 48..51: output projection, 64 vocab columns each.
// Wave tiling (layer): mt0=(w>>2)*2 batch tiles; u=(w&3) unit-group; N-tile g = gate g
// of units n0+u*16.. -> acc[i][0..3][r] are the i/f/g/o gates of one (batch,unit).

__global__ __launch_bounds__(256)
void lstm_step(const int* __restrict__ tokens,
               const h16* __restrict__ Wih0h,   // [G][Vn] row-major (token gather)
               const h16* __restrict__ WhhSwz,  // [3] x swizzled [32][256][32][16]
               const h16* __restrict__ WihSwz,  // [2] x swizzled
               const float* __restrict__ bsum,  // [3][G]
               h16* __restrict__ hh,            // [2 banks][3][Bn][Hn]
               float* __restrict__ cst,         // [3][Bn][Hn]
               h16* __restrict__ htop,          // [Tn][Bn][Hn] (masked top h)
               const int* __restrict__ lengths, // [Bn]
               const h16* __restrict__ WoutSwz, // swizzled [32][16][32][16]
               const float* __restrict__ bout,  // [Vn]
               float* __restrict__ out,         // [Bn][Vn][Tn]
               int k)
{
  const int tid    = threadIdx.x;
  const int w      = tid >> 5;
  const int lane   = tid & 31;
  const int lrow   = lane & 15;
  const int hiHalf = lane >> 4;

  if (blockIdx.x < 48) {
    // ---------------- LSTM cell slice ----------------
    const int l  = blockIdx.x >> 4;
    const int n0 = (blockIdx.x & 15) * 64;
    const int t  = k - l;
    if (t < 0 || t >= Tn) return;

    const h16* hprev = hh + (size_t)(((t-1)&1)*3 + l) * (Bn*Hn);
    h16*       hcurw = hh + (size_t)((t&1)*3 + l) * (Bn*Hn);
    const h16* xcur  = (l > 0) ? hh + (size_t)((t&1)*3 + (l-1)) * (Bn*Hn) : nullptr;

    const int mt0  = (w >> 2) * 2;        // 2 batch tiles per wave
    const int u    = w & 3;               // 16-unit group
    const int nIdx = n0 + u*16 + lrow;    // this lane's hidden unit

    int jt[4];                            // swizzled jtile per gate (of 256)
    #pragma unroll
    for (int g = 0; g < 4; ++g) jt[g] = g*64 + (n0 >> 4) + u;

    v8f acc[2][4];
    #pragma unroll
    for (int i = 0; i < 2; ++i)
      #pragma unroll
      for (int g = 0; g < 4; ++g) acc[i][g] = {};

    const int nPhase = (l == 0) ? 1 : 2;  // layer0 x-path is a token gather
    for (int ph = 0; ph < nPhase; ++ph) {
      const h16* pA = (l == 0) ? hprev : (ph ? hprev : xcur);
      const h16* pB = (l == 0) ? (WhhSwz + (size_t)l*Hn*G)
                               : (ph ? (WhhSwz + (size_t)l*Hn*G)
                                     : (WihSwz + (size_t)(l-1)*Hn*G));
      #pragma unroll 2
      for (int kc = 0; kc < 32; ++kc) {   // K = 32 chunks of 32
        v16h af[2];
        #pragma unroll
        for (int i = 0; i < 2; ++i)
          af[i] = ldfrag(pA + (size_t)((mt0+i)*16 + lrow)*Hn + kc*32 + hiHalf*16, 8);
        if (kc + 1 < 32)
          __builtin_prefetch(pB + ((size_t)(kc+1)*256 + jt[0])*512 + lane*16, 0, 1);
        #pragma unroll
        for (int g = 0; g < 4; ++g) {
          v16h bf = ldfrag(pB + ((size_t)kc*256 + jt[g])*512 + lane*16, 8);
          #pragma unroll
          for (int i = 0; i < 2; ++i)
            acc[i][g] = __builtin_amdgcn_wmma_f32_16x16x32_f16(
                false, af[i], false, bf, (short)0, acc[i][g], false, false);
        }
      }
    }

    // ---------------- in-register LSTM cell epilogue ----------------
    float bs[4];
    #pragma unroll
    for (int g = 0; g < 4; ++g) bs[g] = bsum[l*G + g*Hn + nIdx];

    #pragma unroll
    for (int i = 0; i < 2; ++i)
      #pragma unroll
      for (int r = 0; r < 8; ++r) {
        int m = (mt0 + i)*16 + r + hiHalf*8;       // batch index
        float gi = acc[i][0][r] + bs[0];
        float gf = acc[i][1][r] + bs[1];
        float gg = acc[i][2][r] + bs[2];
        float go = acc[i][3][r] + bs[3];
        if (l == 0) {
          int tk = tokens[t*Bn + m];               // one-hot matmul == gather
          gi += (float)Wih0h[(size_t)nIdx*Vn + tk];
          gf += (float)Wih0h[(size_t)(Hn   + nIdx)*Vn + tk];
          gg += (float)Wih0h[(size_t)(2*Hn + nIdx)*Vn + tk];
          go += (float)Wih0h[(size_t)(3*Hn + nIdx)*Vn + tk];
        }
        float ig = sigf(gi), fg = sigf(gf), og = sigf(go);
        float gv = tanhf(gg);
        size_t cs = ((size_t)l*Bn + m)*Hn + nIdx;
        float c2 = fg*cst[cs] + ig*gv;
        cst[cs] = c2;
        float h2 = og * tanhf(c2);
        hcurw[(size_t)m*Hn + nIdx] = (h16)h2;      // unmasked recurrent state
        if (l == 2) {
          float mk = (t < lengths[m]) ? 1.f : 0.f; // mask only the output path
          htop[((size_t)t*Bn + m)*Hn + nIdx] = (h16)(h2 * mk);
        }
      }
  } else {
    // ---------------- output projection: logits[b][v][t] ----------------
    const int t = k - 3;
    if (t < 0 || t >= Tn) return;
    const int v0 = (blockIdx.x - 48) * 64;
    const h16* pA = htop + (size_t)t*Bn*Hn;

    const int mt  = w >> 1;
    const int vtb = (w & 1) * 2;
    v8f acc[2]; acc[0] = {}; acc[1] = {};

    #pragma unroll 2
    for (int kc = 0; kc < 32; ++kc) {
      v16h af = ldfrag(pA + (size_t)(mt*16 + lrow)*Hn + kc*32 + hiHalf*16, 8);
      #pragma unroll
      for (int j = 0; j < 2; ++j) {
        int jtile = (v0 >> 4) + vtb + j;           // of 16 vocab tiles
        v16h bf = ldfrag(WoutSwz + ((size_t)kc*16 + jtile)*512 + lane*16, 8);
        acc[j] = __builtin_amdgcn_wmma_f32_16x16x32_f16(
            false, af, false, bf, (short)0, acc[j], false, false);
      }
    }
    #pragma unroll
    for (int j = 0; j < 2; ++j) {
      int v = v0 + (vtb + j)*16 + lrow;
      float bo = bout[v];
      #pragma unroll
      for (int r = 0; r < 8; ++r) {
        int b = mt*16 + r + hiHalf*8;
        out[((size_t)b*Vn + v)*Tn + t] = acc[j][r] + bo;
      }
    }
  }
}

// ---------------- host ----------------

extern "C" void kernel_launch(void* const* d_in, const int* in_sizes, int n_in,
                              void* d_out, int out_size, void* d_ws, size_t ws_size,
                              hipStream_t stream) {
  (void)in_sizes; (void)n_in; (void)out_size; (void)ws_size;
  const float* inputs   = (const float*)d_in[0];
  const int*   lengths  = (const int*)d_in[1];
  const float* Wih0     = (const float*)d_in[2];
  const float* Whh0     = (const float*)d_in[3];
  const float* bih0     = (const float*)d_in[4];
  const float* bhh0     = (const float*)d_in[5];
  const float* Wih_rest = (const float*)d_in[6];
  const float* Whh_rest = (const float*)d_in[7];
  const float* bih_rest = (const float*)d_in[8];
  const float* bhh_rest = (const float*)d_in[9];
  const float* Wout     = (const float*)d_in[10];
  const float* boutp    = (const float*)d_in[11];
  float* out = (float*)d_out;

  char* ws = (char*)d_ws;
  size_t off = 0;
  auto carve = [&](size_t bytes) -> void* {
    void* p = ws + off;
    off = (off + bytes + 255) & ~(size_t)255;
    return p;
  };
  int*   tokens  = (int*)  carve((size_t)Tn*Bn*4);
  float* bsum    = (float*)carve((size_t)3*G*4);
  h16*   Wih0h   = (h16*)  carve((size_t)G*Vn*2);
  h16*   WhhSwz  = (h16*)  carve((size_t)3*Hn*G*2);
  h16*   WihSwz  = (h16*)  carve((size_t)2*Hn*G*2);
  h16*   WoutSwz = (h16*)  carve((size_t)Hn*Vn*2);
  h16*   hhbuf   = (h16*)  carve((size_t)2*3*Bn*Hn*2);
  float* cstb    = (float*)carve((size_t)3*Bn*Hn*4);
  h16*   htop    = (h16*)  carve((size_t)Tn*Bn*Hn*2);

  auto grid = [](int n){ return (n + 255)/256; };
  const int nswzW = (Hn/32)*(G/16)*32;   // threads per Whh/Wih swizzle
  const int nswzO = (Hn/32)*(Vn/16)*32;

  k_tokens<<<grid(Tn*Bn), 256, 0, stream>>>(inputs, tokens, Tn*Bn);
  k_conv  <<<grid(G*Vn), 256, 0, stream>>>(Wih0, Wih0h, G*Vn);
  k_swz   <<<grid(nswzW), 256, 0, stream>>>(Whh0, WhhSwz, G, Hn);
  k_swz   <<<grid(nswzW), 256, 0, stream>>>(Whh_rest,                WhhSwz + (size_t)1*Hn*G, G, Hn);
  k_swz   <<<grid(nswzW), 256, 0, stream>>>(Whh_rest + (size_t)G*Hn, WhhSwz + (size_t)2*Hn*G, G, Hn);
  k_swz   <<<grid(nswzW), 256, 0, stream>>>(Wih_rest,                WihSwz,                  G, Hn);
  k_swz   <<<grid(nswzW), 256, 0, stream>>>(Wih_rest + (size_t)G*Hn, WihSwz + (size_t)1*Hn*G, G, Hn);
  k_swz   <<<grid(nswzO), 256, 0, stream>>>(Wout, WoutSwz, Vn, Hn);
  k_bsum  <<<grid(3*G), 256, 0, stream>>>(bih0, bhh0, bih_rest, bhh_rest, bsum);
  int nh = 2*3*Bn*Hn, nc = 3*Bn*Hn;
  k_zero  <<<grid(nh), 256, 0, stream>>>(hhbuf, cstb, nh, nc);

  // wavefront over (time, layer): each launch = 3 independent layer GEMMs + projection
  for (int k = 0; k < Tn + 3; ++k)
    lstm_step<<<52, 256, 0, stream>>>(tokens, Wih0h, WhhSwz, WihSwz, bsum, hhbuf, cstb,
                                      htop, lengths, WoutSwz, boutp, out, k);
}
